// RetinaNet_48507360641525
// MI455X (gfx1250) — matrix operand order repaired and checked by reading, not compiled, not run on hardware
//
#include <hip/hip_runtime.h>
#include <hip/hip_bf16.h>

typedef unsigned short u16;
typedef unsigned int   u32;
typedef __attribute__((ext_vector_type(16))) __bf16 v16bf;
typedef __attribute__((ext_vector_type(8)))  float  v8f;
typedef __attribute__((ext_vector_type(4)))  int    v4i;

#define ANCH 76725
#define NEGINF -3.2e38f

#if defined(__has_builtin)
# if __has_builtin(__builtin_amdgcn_global_load_async_to_lds_b128)
#  define HAVE_ASYNC_LDS 1
# endif
# if __has_builtin(__builtin_amdgcn_s_wait_asynccnt)
#  define HAVE_WAIT_ASYNC 1
# endif
#endif

__device__ __forceinline__ u16 f2bf(float f) {
    u32 u = __float_as_uint(f);
    u32 r = (u + 0x7FFFu + ((u >> 16) & 1u)) >> 16;   // round-nearest-even
    return (u16)r;
}
__device__ __forceinline__ float bf2f(u16 s) {
    return __uint_as_float(((u32)s) << 16);
}

union Frag { v16bf v; uint4 q[2]; };

// 16-byte global -> LDS copy: async DMA (ASYNCcnt) when available, else
// load + ds_store (ordered by the same workgroup barriers).
__device__ __forceinline__ void cp16(const u16* g, u16* l) {
#ifdef HAVE_ASYNC_LDS
    __builtin_amdgcn_global_load_async_to_lds_b128(
        (__attribute__((address_space(1))) v4i*)(u16*)g,
        (__attribute__((address_space(3))) v4i*)l, 0, 0);
#else
    *(uint4*)l = *(const uint4*)g;
#endif
}
__device__ __forceinline__ void wait_async() {
#ifdef HAVE_ASYNC_LDS
# ifdef HAVE_WAIT_ASYNC
    __builtin_amdgcn_s_wait_asynccnt(0);
# else
    asm volatile("s_wait_asynccnt 0x0" ::: "memory");
# endif
#endif
}

// ---------------------------------------------------------------------------
// Weight repack: [L][oc][ic][3][3] f32  ->  [L][oc][tap][ic] bf16
// ---------------------------------------------------------------------------
__global__ void repack_weights(const float* __restrict__ src, u16* __restrict__ dst, int total) {
    int idx = blockIdx.x * blockDim.x + threadIdx.x;
    if (idx >= total) return;
    int ic   = idx & 255;
    int tap  = (idx >> 8) % 9;
    int rest = idx / (256 * 9);          // rest = l*256 + oc
    dst[idx] = f2bf(src[((long)rest * 256 + ic) * 9 + tap]);
}

// ---------------------------------------------------------------------------
// NCHW f32 -> padded NHWC bf16 (zero border), whole padded buffer written
// ---------------------------------------------------------------------------
__global__ void feat_to_padded(const float* __restrict__ src, u16* __restrict__ dst, int h, int w) {
    int wp = w + 2;
    int padN = (h + 2) * wp * 256;
    int idx = blockIdx.x * blockDim.x + threadIdx.x;
    if (idx >= padN) return;
    int c = idx & 255;
    int cell = idx >> 8;
    int px = cell % wp, py = cell / wp;
    u16 v = 0;
    if (px >= 1 && px <= w && py >= 1 && py <= h)
        v = f2bf(src[((long)c * h + (py - 1)) * w + (px - 1)]);
    dst[idx] = v;
}

__global__ void zero_bf16(u16* __restrict__ p, int n) {
    int i = blockIdx.x * blockDim.x + threadIdx.x;
    if (i < n) p[i] = 0;
}

// ---------------------------------------------------------------------------
// Implicit-GEMM 3x3 conv, 256->256, bias+ReLU, bf16 WMMA (f32 accum),
// LDS-staged and double-buffered:
//   block tile 64(M) x 128(N), 8 waves = 4 M-subtiles x 2 N-subtiles,
//   K loop: 36 chunks of 64 (9 taps x 4). Per chunk the block async-copies
//   A(64x64) + B(128x64) bf16 into ping/pong LDS (48 KB), waits ASYNCcnt,
//   barriers, then each wave runs 8 WMMAs off ds_load_b128 fragments while
//   the DMA for the next chunk is in flight. gridDim.z selects head.
// ---------------------------------------------------------------------------
#define WMMA4(AF, B0, B1, B2, B3)                                                                   \
    acc[0] = __builtin_amdgcn_wmma_f32_16x16x32_bf16(false, AF.v, false, B0.v, (short)0, acc[0], false, false); \
    acc[1] = __builtin_amdgcn_wmma_f32_16x16x32_bf16(false, AF.v, false, B1.v, (short)0, acc[1], false, false); \
    acc[2] = __builtin_amdgcn_wmma_f32_16x16x32_bf16(false, AF.v, false, B2.v, (short)0, acc[2], false, false); \
    acc[3] = __builtin_amdgcn_wmma_f32_16x16x32_bf16(false, AF.v, false, B3.v, (short)0, acc[3], false, false);

#define COPYBUF(BUF, S)                                                          \
    {                                                                            \
        const int  tp_ = (S) >> 2;                                               \
        const int  kq_ = ((S) & 3) * 64;                                         \
        const int  dy_ = tp_ / 3 - 1;                                            \
        const int  dx_ = tp_ - (tp_ / 3) * 3 - 1;                                \
        const long ao_ = ((long)dy_ * wp + dx_) * 256 + kq_;                     \
        const long k0_ = (long)tp_ * 256 + kq_;                                  \
        cp16(srcp + aRow0 + ao_, &Ash[BUF][ar0][aq]);                            \
        cp16(srcp + aRow1 + ao_, &Ash[BUF][ar1][aq]);                            \
        cp16(wbp + bg + k0_,             &Bsh[BUF][boc     ][aq]);               \
        cp16(wbp + bg + 32 * 2304 + k0_, &Bsh[BUF][boc + 32][aq]);               \
        cp16(wbp + bg + 64 * 2304 + k0_, &Bsh[BUF][boc + 64][aq]);               \
        cp16(wbp + bg + 96 * 2304 + k0_, &Bsh[BUF][boc + 96][aq]);               \
    }

#define FRAGMMA(CUR, KS)                                                         \
    {                                                                            \
        Frag a, b0, b1, b2, b3;                                                  \
        a.q[0]  = *(const uint4*)&Ash[CUR][arow][(KS) * 32 + kbA];               \
        a.q[1]  = *(const uint4*)&Ash[CUR][arow][(KS) * 32 + kbA + 8];           \
        b0.q[0] = *(const uint4*)&Bsh[CUR][brow +  0][(KS) * 32 + kbB];          \
        b0.q[1] = *(const uint4*)&Bsh[CUR][brow +  0][(KS) * 32 + kbB + 8];      \
        b1.q[0] = *(const uint4*)&Bsh[CUR][brow + 16][(KS) * 32 + kbB];          \
        b1.q[1] = *(const uint4*)&Bsh[CUR][brow + 16][(KS) * 32 + kbB + 8];      \
        b2.q[0] = *(const uint4*)&Bsh[CUR][brow + 32][(KS) * 32 + kbB];          \
        b2.q[1] = *(const uint4*)&Bsh[CUR][brow + 32][(KS) * 32 + kbB + 8];      \
        b3.q[0] = *(const uint4*)&Bsh[CUR][brow + 48][(KS) * 32 + kbB];          \
        b3.q[1] = *(const uint4*)&Bsh[CUR][brow + 48][(KS) * 32 + kbB + 8];      \
        WMMA4(a, b0, b1, b2, b3);                                                \
    }

__global__ __launch_bounds__(256) void conv256_wmma_lds(
    const u16* __restrict__ srcC, const u16* __restrict__ srcR,
    u16* __restrict__ dstC, u16* __restrict__ dstR,
    const u16* __restrict__ wbC, const u16* __restrict__ wbR,
    const float* __restrict__ biasC, const float* __restrict__ biasR,
    int h, int w)
{
    __shared__ u16 Ash[2][64][64];     // 16 KB
    __shared__ u16 Bsh[2][128][64];    // 32 KB

    const int head = blockIdx.z;
    const u16*   srcp  = head ? srcR  : srcC;
    u16*         dstp  = head ? dstR  : dstC;
    const u16*   wbp   = head ? wbR   : wbC;
    const float* biasp = head ? biasR : biasC;

    const int hw = h * w, wp = w + 2;
    const int t    = threadIdx.x;
    const int lane = t & 31, wave = t >> 5;
    const int tileM0 = blockIdx.x * 64;
    const int tileN0 = blockIdx.y * 128;
    const int mw = wave & 3, nw = wave >> 2;
    const int lhalf = (lane < 16) ? 0 : 1;
    const int l15   = lane & 15;
    const int kbA = lhalf * 16;        // A frag: lanes16-31 hold K+16
    const int kbB = lhalf * 16;        // B frag: lanes16-31 hold K+16
    const int arow = mw * 16 + l15;    // A row in LDS tile
    const int brow = nw * 64 + l15;    // B row (oc) in LDS tile

    // ---- cooperative-copy descriptors (per thread) ----
    const int aq  = (t & 7) * 8;       // 16B sub-chunk within a 64-elem row
    const int ar0 = t >> 3;            // A rows: t>>3 and +32  (64 rows total)
    const int ar1 = ar0 + 32;
    long aRow0, aRow1;
    {
        int am0 = tileM0 + ar0; if (am0 >= hw) am0 = hw - 1;
        int y0 = am0 / w, x0 = am0 - y0 * w;
        aRow0 = (((long)(y0 + 1) * wp) + (x0 + 1)) * 256 + aq;
        int am1 = tileM0 + ar1; if (am1 >= hw) am1 = hw - 1;
        int y1 = am1 / w, x1 = am1 - y1 * w;
        aRow1 = (((long)(y1 + 1) * wp) + (x1 + 1)) * 256 + aq;
    }
    const int  boc = t >> 3;                              // B oc rows: boc + {0,32,64,96}
    const long bg  = (long)(tileN0 + boc) * 2304 + aq;

    v8f acc[4] = {v8f{}, v8f{}, v8f{}, v8f{}};

    COPYBUF(0, 0);                       // prologue: chunk 0 -> buffer 0
#pragma unroll
    for (int s = 0; s < 36; ++s) {
        const int cur = s & 1;
        wait_async();                    // my copies for `cur` complete
        __syncthreads();                 // everyone's copies visible; prev reads done
        if (s + 1 < 36) { COPYBUF(cur ^ 1, s + 1); }   // DMA next chunk in flight
        FRAGMMA(cur, 0);                 // 2 sub-chunks of K=32 -> 8 WMMAs
        FRAGMMA(cur, 1);
    }

    // --- store D (bias + ReLU, bf16, interior only: border stays zero) ---
    const int n0 = tileN0 + nw * 64 + l15;
    __builtin_prefetch(biasp + n0, 0, 0);          // global_prefetch_b8
#pragma unroll
    for (int f = 0; f < 4; ++f) {
        const int n = n0 + f * 16;
        const float bv = biasp[n];
#pragma unroll
        for (int r = 0; r < 8; ++r) {
            const int mm = tileM0 + mw * 16 + r + lhalf * 8;   // D: VGPR r -> row r / r+8
            if (mm < hw) {
                const int yy = mm / w, xx = mm - yy * w;
                float v = acc[f][r] + bv;
                v = v > 0.f ? v : 0.f;
                dstp[(((long)(yy + 1) * wp) + (xx + 1)) * 256 + n] = f2bf(v);
            }
        }
    }
}

// ---------------------------------------------------------------------------
// Output convs (small N): plain fp32 dot products (~2% of total FLOPs)
// ---------------------------------------------------------------------------
__global__ void outconv_cls(const u16* __restrict__ act, const float* __restrict__ wgt,
                            const float* __restrict__ bias, float* __restrict__ scores,
                            int h, int w, int abase)
{
    int hw = h * w, wp = w + 2;
    int idx = blockIdx.x * blockDim.x + threadIdx.x;
    if (idx >= hw * 9) return;
    int m = idx / 9, ch = idx - m * 9;
    int y = m / w, x = m - y * w;
    float acc = bias[ch];
    for (int tap = 0; tap < 9; ++tap) {
        int dy = tap / 3 - 1, dx = tap % 3 - 1;
        const u16*   ap = act + (((long)(y + 1 + dy) * wp) + (x + 1 + dx)) * 256;
        const float* wq = wgt + (long)ch * 2304 + tap;      // [oc][ic][ky][kx]
        for (int ic = 0; ic < 256; ++ic) acc += bf2f(ap[ic]) * wq[ic * 9];
    }
    scores[abase + m * 9 + ch] = 1.f / (1.f + __expf(-acc));   // sigmoid fused
}

__global__ void outconv_reg(const u16* __restrict__ act, const float* __restrict__ wgt,
                            const float* __restrict__ bias, float* __restrict__ regout,
                            int h, int w, int abase)
{
    int hw = h * w, wp = w + 2;
    int idx = blockIdx.x * blockDim.x + threadIdx.x;
    if (idx >= hw * 36) return;
    int m = idx / 36, ch = idx - m * 36;
    int y = m / w, x = m - y * w;
    float acc = bias[ch];
    for (int tap = 0; tap < 9; ++tap) {
        int dy = tap / 3 - 1, dx = tap % 3 - 1;
        const u16*   ap = act + (((long)(y + 1 + dy) * wp) + (x + 1 + dx)) * 256;
        const float* wq = wgt + (long)ch * 2304 + tap;
        for (int ic = 0; ic < 256; ++ic) acc += bf2f(ap[ic]) * wq[ic * 9];
    }
    // NHWC flatten: (m,36) -> (m*9 anchors, 4 coords): flat idx = (abase+m*9)*4 + ch
    regout[((long)(abase + m * 9)) * 4 + ch] = acc;
}

// ---------------------------------------------------------------------------
// Anchor generation + box decode + clip + score threshold
// ---------------------------------------------------------------------------
__global__ void decode_kernel(const float* __restrict__ scores, const float* __restrict__ reg,
                              float* __restrict__ boxes, float* __restrict__ swork)
{
    int i = blockIdx.x * blockDim.x + threadIdx.x;
    if (i >= ANCH) return;
    const int bases[6] = {0, 57600, 72000, 75600, 76500, 76725};
    const int fws[5]   = {80, 40, 20, 10, 5};
    int lvl = 0;
    while (lvl < 4 && i >= bases[lvl + 1]) ++lvl;
    int li = i - bases[lvl];
    int cell = li / 9, a = li - cell * 9;
    int fw = fws[lvl];
    int yy = cell / fw, xx = cell - yy * fw;
    float stride = (float)(8 << lvl);
    float basesz = (float)(32 << lvl);
    int ri = a / 3, si = a - ri * 3;
    float ratio = (ri == 0) ? 0.5f : ((ri == 1) ? 1.0f : 2.0f);
    float scale = exp2f((float)si / 3.0f);
    float sz = basesz * scale;
    float aw = sz * rsqrtf(ratio);         // sqrt(area/ratio)
    float ah = aw * ratio;
    float cx = (xx + 0.5f) * stride, cy = (yy + 0.5f) * stride;
    float r0 = reg[4 * i], r1 = reg[4 * i + 1], r2 = reg[4 * i + 2], r3 = reg[4 * i + 3];
    float pcx = cx + r0 * 0.1f * aw;
    float pcy = cy + r1 * 0.1f * ah;
    float pw  = __expf(r2 * 0.2f) * aw;
    float ph  = __expf(r3 * 0.2f) * ah;
    float x1 = fminf(fmaxf(pcx - 0.5f * pw, 0.f), 640.f);
    float y1 = fminf(fmaxf(pcy - 0.5f * ph, 0.f), 640.f);
    float x2 = fminf(fmaxf(pcx + 0.5f * pw, 0.f), 640.f);
    float y2 = fminf(fmaxf(pcy + 0.5f * ph, 0.f), 640.f);
    boxes[4 * i] = x1; boxes[4 * i + 1] = y1; boxes[4 * i + 2] = x2; boxes[4 * i + 3] = y2;
    float s = scores[i];
    swork[i] = (s > 0.05f) ? s : NEGINF;
}

// ---------------------------------------------------------------------------
// Greedy NMS: single block, 300 iterations of argmax + suppression.
// out: [300 scores][300 labels][300*4 boxes]
// ---------------------------------------------------------------------------
__global__ __launch_bounds__(1024) void nms_kernel(const float* __restrict__ boxes,
                                                   float* __restrict__ s,
                                                   float* __restrict__ out)
{
    __shared__ float sv[1024];
    __shared__ int   si[1024];
    __shared__ float bsh[4];
    const int tid = threadIdx.x;
    for (int it = 0; it < 300; ++it) {
        float best = NEGINF * 2.f; int bi = 0;
        for (int k = tid; k < ANCH; k += 1024) {
            float v = s[k];
            if (v > best) { best = v; bi = k; }
        }
        sv[tid] = best; si[tid] = bi;
        __syncthreads();
        for (int off = 512; off > 0; off >>= 1) {
            if (tid < off && sv[tid + off] > sv[tid]) { sv[tid] = sv[tid + off]; si[tid] = si[tid + off]; }
            __syncthreads();
        }
        const int j = si[0];
        const float sj = sv[0];
        if (tid == 0) {
            bsh[0] = boxes[4 * j];     bsh[1] = boxes[4 * j + 1];
            bsh[2] = boxes[4 * j + 2]; bsh[3] = boxes[4 * j + 3];
        }
        __syncthreads();
        const float ax1 = bsh[0], ay1 = bsh[1], ax2 = bsh[2], ay2 = bsh[3];
        const float areaj = (ax2 - ax1) * (ay2 - ay1);
        for (int k = tid; k < ANCH; k += 1024) {
            float x1 = boxes[4 * k], y1 = boxes[4 * k + 1];
            float x2 = boxes[4 * k + 2], y2 = boxes[4 * k + 3];
            float inter = fmaxf(fminf(ax2, x2) - fmaxf(ax1, x1), 0.f) *
                          fmaxf(fminf(ay2, y2) - fmaxf(ay1, y1), 0.f);
            float iou = inter / ((x2 - x1) * (y2 - y1) + areaj - inter + 1e-8f);
            if (iou > 0.5f) s[k] = NEGINF;   // includes j itself (self-IoU == 1)
        }
        if (tid == 0) {
            bool keep = sj > NEGINF * 0.5f;
            out[it]             = keep ? sj : 0.f;
            out[300 + it]       = keep ? 0.f : -1.f;   // label 0 or -1
            out[600 + 4 * it]     = keep ? bsh[0] : 0.f;
            out[600 + 4 * it + 1] = keep ? bsh[1] : 0.f;
            out[600 + 4 * it + 2] = keep ? bsh[2] : 0.f;
            out[600 + 4 * it + 3] = keep ? bsh[3] : 0.f;
        }
        __threadfence_block();
        __syncthreads();
    }
}

// ---------------------------------------------------------------------------
extern "C" void kernel_launch(void* const* d_in, const int* in_sizes, int n_in,
                              void* d_out, int out_size, void* d_ws, size_t ws_size,
                              hipStream_t stream)
{
    const float* feats[5] = {(const float*)d_in[1], (const float*)d_in[2],
                             (const float*)d_in[3], (const float*)d_in[4],
                             (const float*)d_in[5]};
    const float* cls_w  = (const float*)d_in[6];
    const float* cls_b  = (const float*)d_in[7];
    const float* cls_ow = (const float*)d_in[8];
    const float* cls_ob = (const float*)d_in[9];
    const float* reg_w  = (const float*)d_in[10];
    const float* reg_b  = (const float*)d_in[11];
    const float* reg_ow = (const float*)d_in[12];
    const float* reg_ob = (const float*)d_in[13];
    float* out = (float*)d_out;

    // ---- workspace carve-up ----
    const long PADMAX = 82L * 82L * 256L;          // largest padded NHWC level
    const long WELEM  = 4L * 256L * 9L * 256L;     // 4 layers of repacked weights
    char* ws = (char*)d_ws;
    size_t off = 0;
    auto carve = [&](size_t bytes) { char* p = ws + off; off = (off + bytes + 255) & ~(size_t)255; return p; };
    u16*   buf0   = (u16*)carve(PADMAX * 2);       // converted input (shared by heads)
    u16*   c1     = (u16*)carve(PADMAX * 2);       // cls ping-pong
    u16*   c2     = (u16*)carve(PADMAX * 2);
    u16*   r1     = (u16*)carve(PADMAX * 2);       // reg ping-pong
    u16*   r2     = (u16*)carve(PADMAX * 2);
    u16*   wbcls  = (u16*)carve(WELEM * 2);
    u16*   wbreg  = (u16*)carve(WELEM * 2);
    float* scores = (float*)carve(ANCH * 4);
    float* swork  = (float*)carve(ANCH * 4);
    float* regout = (float*)carve((size_t)ANCH * 4 * 4);
    float* boxes  = (float*)carve((size_t)ANCH * 4 * 4);

    // ---- repack shared-conv weights to bf16 [l][oc][tap][ic] ----
    {
        int total = (int)WELEM;
        int g = (total + 255) / 256;
        repack_weights<<<g, 256, 0, stream>>>(cls_w, wbcls, total);
        repack_weights<<<g, 256, 0, stream>>>(reg_w, wbreg, total);
    }

    const int HS[5]    = {80, 40, 20, 10, 5};
    const int abase[5] = {0, 57600, 72000, 75600, 76500};

    for (int lvl = 0; lvl < 5; ++lvl) {
        const int h = HS[lvl], w = HS[lvl];
        const int hw = h * w;
        const int padN = (h + 2) * (w + 2) * 256;
        const int gpad = (padN + 255) / 256;

        feat_to_padded<<<gpad, 256, 0, stream>>>(feats[lvl], buf0, h, w);
        zero_bf16<<<gpad, 256, 0, stream>>>(c1, padN);   // establish zero borders
        zero_bf16<<<gpad, 256, 0, stream>>>(c2, padN);
        zero_bf16<<<gpad, 256, 0, stream>>>(r1, padN);
        zero_bf16<<<gpad, 256, 0, stream>>>(r2, padN);

        dim3 cg((hw + 63) / 64, 2, 2);                   // z: cls / reg head
        u16* chC[5] = {buf0, c1, c2, c1, c2};            // buf0 preserved
        u16* chR[5] = {buf0, r1, r2, r1, r2};

        for (int l = 0; l < 4; ++l)
            conv256_wmma_lds<<<cg, 256, 0, stream>>>(
                chC[l], chR[l], chC[l + 1], chR[l + 1],
                wbcls + (long)l * 589824, wbreg + (long)l * 589824,
                cls_b + l * 256, reg_b + l * 256, h, w);

        outconv_cls<<<(hw * 9 + 255) / 256, 256, 0, stream>>>(c2, cls_ow, cls_ob,
                                                              scores, h, w, abase[lvl]);
        outconv_reg<<<(hw * 36 + 255) / 256, 256, 0, stream>>>(r2, reg_ow, reg_ob,
                                                               regout, h, w, abase[lvl]);
    }

    decode_kernel<<<(ANCH + 255) / 256, 256, 0, stream>>>(scores, regout, boxes, swork);
    nms_kernel<<<1, 1024, 0, stream>>>(boxes, swork, out);
}
